// SyllaBERT_52003464020012
// MI455X (gfx1250) — compile-verified, matrix operand order checked by reading, not run on hardware
//
#include <hip/hip_runtime.h>
#include <hip/hip_bf16.h>

typedef __attribute__((ext_vector_type(16))) _Float16 v16h;
typedef __attribute__((ext_vector_type(8)))  float    v8f;
typedef __attribute__((ext_vector_type(4)))  unsigned int v4u;
typedef __attribute__((ext_vector_type(8)))  int v8i;
typedef __attribute__((ext_vector_type(4)))  int v4i;

#if defined(__gfx1250__) && __has_builtin(__builtin_amdgcn_tensor_load_to_lds)
#define HAVE_TDM 1
#else
#define HAVE_TDM 0
#endif

#if HAVE_TDM
// Tensor Data Mover: DMA a 3D tile (X=tile0 contiguous elems, Y=tile1 rows stride
// stride0, Z=tile2 planes stride stride1; 2-byte elements) from global to LDS.
// D# layout per CDNA5 ISA sect 8.3/8.4. Issued by one wave,
// completion via TENSORcnt.
__device__ __forceinline__ void tdm_load_2b(unsigned long long gaddr, unsigned lds_addr,
                                            unsigned tile0, unsigned tile1, unsigned tile2,
                                            unsigned long long stride0,
                                            unsigned long long stride1,
                                            unsigned tdim0, unsigned tdim1) {
  v4u g0;
  g0.x = 1u;                                            // count=1, user descriptor
  g0.y = lds_addr;                                      // lds_addr [63:32]
  g0.z = (unsigned)(gaddr & 0xffffffffu);               // global_addr [95:64]
  g0.w = (unsigned)((gaddr >> 32) & 0x1ffffffu) | (2u << 30);  // addr[56:32] | type=2
  v8i g1;
  g1[0] = (int)(1u << 16);                              // data_size=1 -> 2 bytes
  g1[1] = (int)((tdim0 & 0xffffu) << 16);               // tensor_dim0[15:0]
  g1[2] = (int)(((tdim0 >> 16) & 0xffffu) | ((tdim1 & 0xffffu) << 16));
  g1[3] = (int)(((tdim1 >> 16) & 0xffffu) | ((tile0 & 0xffffu) << 16));
  g1[4] = (int)((tile1 & 0xffffu) | ((tile2 & 0xffffu) << 16));
  g1[5] = (int)(stride0 & 0xffffffffu);
  g1[6] = (int)(((stride0 >> 32) & 0xffffu) | ((stride1 & 0xffffu) << 16));
  g1[7] = (int)((stride1 >> 16) & 0xffffffffu);
  v4i z4 = {0, 0, 0, 0};
#if __clang_major__ >= 23
  v8i z8 = {0, 0, 0, 0, 0, 0, 0, 0};
  __builtin_amdgcn_tensor_load_to_lds(g0, g1, z4, z4, z8, 0);
#else
  __builtin_amdgcn_tensor_load_to_lds(g0, g1, z4, z4, 0);
#endif
#if __has_builtin(__builtin_amdgcn_s_wait_tensorcnt)
  __builtin_amdgcn_s_wait_tensorcnt(0);
#endif
}
__device__ __forceinline__ unsigned lds_addr_of(const void* p) {
  return (unsigned)(unsigned long long)(__attribute__((address_space(3))) const void*)p;
}
#endif

// ---------------------------------------------------------------------------
// Conv layer 0: in_c=1, k=10, s=5  (K too small for WMMA -> VALU kernel)
// ---------------------------------------------------------------------------
__global__ void conv0_kernel(const float* __restrict__ x, const float* __restrict__ w,
                             const float* __restrict__ bias, _Float16* __restrict__ out) {
  const int t = blockIdx.x;            // 0..31998
  const int b = blockIdx.y;
  const float* xs = x + (size_t)b * 160000 + (size_t)t * 5;
  float s[10];
#pragma unroll
  for (int j = 0; j < 10; ++j) s[j] = xs[j];
  for (int co = threadIdx.x; co < 512; co += blockDim.x) {
    float acc = bias[co];
#pragma unroll
    for (int j = 0; j < 10; ++j) acc = fmaf(w[co * 10 + j], s[j], acc);
    acc = fmaxf(acc, 0.f);
    out[((size_t)b * 31999 + t) * 512 + co] = (_Float16)acc;
  }
}

// Rearrange conv weights [oc][ic][k] f32 -> [k][oc][ic] f16
__global__ void convw_cast_kernel(const float* __restrict__ w, _Float16* __restrict__ o,
                                  int oc, int ic, int k) {
  int i = blockIdx.x * blockDim.x + threadIdx.x;
  int n = oc * ic * k;
  if (i >= n) return;
  int j = i % k;
  int ci = (i / k) % ic;
  int co = i / (k * ic);
  o[(size_t)j * oc * ic + (size_t)co * ic + ci] = (_Float16)w[i];
}

__global__ void cast_f32_f16_kernel(const float* __restrict__ in, _Float16* __restrict__ o, int n) {
  int i = blockIdx.x * blockDim.x + threadIdx.x;
  if (i < n) o[i] = (_Float16)in[i];
}

// ---------------------------------------------------------------------------
// WMMA conv. A = weights (M=co, K=ci) staged in LDS via TDM once per block
// (shared by all 8 waves); B = activations (K=ci, N=t) streamed from global.
// Each wave: 16 co x 64 t strip = 4 independent accumulators. Cin==512 always.
// ---------------------------------------------------------------------------
__global__ void conv_wmma_kernel(const _Float16* __restrict__ act,  // [B,Lin,Cin]
                                 const _Float16* __restrict__ wgt,  // [k,Cout,Cin]
                                 const float* __restrict__ bias,
                                 _Float16* __restrict__ out,        // [B,Lout,Cout]
                                 int Lin, int Lout, int Cin, int Cout, int k, int s) {
  __shared__ _Float16 wsm[3 * 16 * 512];   // [k][16 co][512 ci], <= 48 KB
  const int lane = threadIdx.x & 31;
  const int wave = threadIdx.x >> 5;
  const int t0 = (blockIdx.x * 8 + wave) * 64;     // 64 outputs per wave
  const int co0 = blockIdx.y * 16;
  const int b = blockIdx.z;

  // ---- stage weight tile [k][co0..co0+15][0..Cin) -> LDS ----
#if HAVE_TDM
  if (threadIdx.x < 32) {
    tdm_load_2b((unsigned long long)(const void*)(wgt + (size_t)co0 * Cin),
                lds_addr_of(wsm),
                (unsigned)Cin, 16u, (unsigned)k,
                (unsigned long long)Cin, (unsigned long long)Cout * Cin,
                (unsigned)Cin, (unsigned)Cout);
  }
  __syncthreads();
#else
  for (int i = threadIdx.x; i < k * 16 * Cin; i += blockDim.x) {
    int j = i / (16 * Cin);
    int r = (i / Cin) & 15;
    int c = i % Cin;
    wsm[i] = wgt[((size_t)j * Cout + co0 + r) * Cin + c];
  }
  __syncthreads();
#endif
  if (t0 >= Lout) return;

  const int khalf = (lane >> 4) * 16;   // lanes 0-15: K 0..15 ; lanes 16-31: K 16..31
  const int arow = lane & 15;           // co within tile (A-fragment row)
  const int bcol = lane & 15;           // t within sub-tile (B-fragment column)
  const _Float16* actB = act + (size_t)b * Lin * Cin;

  size_t trow[4];
#pragma unroll
  for (int i = 0; i < 4; ++i) {
    int t = t0 + i * 16 + bcol;
    if (t >= Lout) t = Lout - 1;        // clamp loads; stores guarded below
    trow[i] = (size_t)t * s;
  }
  v8f acc[4];
#pragma unroll
  for (int i = 0; i < 4; ++i) acc[i] = (v8f){0.f, 0.f, 0.f, 0.f, 0.f, 0.f, 0.f, 0.f};

  for (int j = 0; j < k; ++j) {
    const _Float16* Aj = wsm + ((size_t)j * 16 + arow) * Cin + khalf;   // LDS
    const _Float16* Bj[4];
#pragma unroll
    for (int i = 0; i < 4; ++i) Bj[i] = actB + (trow[i] + j) * Cin + khalf;
#pragma unroll 2
    for (int k0 = 0; k0 < Cin; k0 += 32) {
      v16h a = *(const v16h*)(Aj + k0);                 // ds_load
      __builtin_prefetch(Bj[0] + k0 + 256, 0, 1);       // global_prefetch_b8
#pragma unroll
      for (int i = 0; i < 4; ++i) {
        v16h bf = *(const v16h*)(Bj[i] + k0);
        acc[i] = __builtin_amdgcn_wmma_f32_16x16x32_f16(false, a, false, bf, (short)0,
                                                        acc[i], false, false);
      }
    }
  }
  // D layout: lane -> N (=t), VGPR r -> M (=co), lanes 16-31 add 8 to M
  const int col = lane & 15;
  const int rbase = (lane >> 4) * 8;
#pragma unroll
  for (int i = 0; i < 4; ++i) {
    const int tt = t0 + i * 16 + col;
    if (tt >= Lout) continue;
#pragma unroll
    for (int r = 0; r < 8; ++r) {
      int co = co0 + rbase + r;
      float v = fmaxf(acc[i][r] + bias[co], 0.f);
      out[((size_t)b * Lout + tt) * Cout + co] = (_Float16)v;
    }
  }
}

// ---------------------------------------------------------------------------
// WMMA linear: out[m][n] = act(bias[n] + sum_k A[m][k]*W[n][k])
// A-token tile [16 x K] staged in LDS via TDM once per block (shared by 8 waves);
// W rows streamed from global. Each wave: 16 m x 64 n strip, 4 accumulators.
// ---------------------------------------------------------------------------
__global__ void linear_wmma_kernel(const _Float16* __restrict__ A,  // [M,K]
                                   const _Float16* __restrict__ W,  // [N,K]
                                   const float* __restrict__ bias,
                                   float* __restrict__ outF, _Float16* __restrict__ outH,
                                   int M, int N, int K, int relu) {
  __shared__ _Float16 asmem[16 * 2048];    // [16 m][K<=2048], <= 64 KB
  const int lane = threadIdx.x & 31;
  const int wave = threadIdx.x >> 5;
  const int n0 = (blockIdx.x * 8 + wave) * 64;
  const int m0 = blockIdx.y * 16;

#if HAVE_TDM
  if (threadIdx.x < 32) {
    tdm_load_2b((unsigned long long)(const void*)(A + (size_t)m0 * K),
                lds_addr_of(asmem),
                (unsigned)K, 16u, 1u,
                (unsigned long long)K, (unsigned long long)K * 16,
                (unsigned)K, (unsigned)M);
  }
  __syncthreads();
#else
  for (int i = threadIdx.x; i < 16 * K; i += blockDim.x) {
    int r = i / K;
    int c = i % K;
    asmem[i] = A[((size_t)m0 + r) * K + c];
  }
  __syncthreads();
#endif
  if (n0 >= N) return;

  const int khalf = (lane >> 4) * 16;
  const _Float16* Ap = asmem + (size_t)(lane & 15) * K + khalf;   // LDS

  const _Float16* Bp[4];
#pragma unroll
  for (int i = 0; i < 4; ++i) {
    int bn = n0 + i * 16 + (lane & 15);
    if (bn >= N) bn = N - 1;
    Bp[i] = W + (size_t)bn * K + khalf;
  }
  v8f acc[4];
#pragma unroll
  for (int i = 0; i < 4; ++i) acc[i] = (v8f){0.f, 0.f, 0.f, 0.f, 0.f, 0.f, 0.f, 0.f};

#pragma unroll 2
  for (int k0 = 0; k0 < K; k0 += 32) {
    v16h a = *(const v16h*)(Ap + k0);                   // ds_load
    __builtin_prefetch(Bp[0] + k0 + 256, 0, 1);
#pragma unroll
    for (int i = 0; i < 4; ++i) {
      v16h b = *(const v16h*)(Bp[i] + k0);
      acc[i] = __builtin_amdgcn_wmma_f32_16x16x32_f16(false, a, false, b, (short)0,
                                                      acc[i], false, false);
    }
  }
  const int col = lane & 15;
  const int rbase = (lane >> 4) * 8;
#pragma unroll
  for (int i = 0; i < 4; ++i) {
    const int n = n0 + i * 16 + col;
    if (n >= N) continue;
    const float bv = bias ? bias[n] : 0.f;
#pragma unroll
    for (int r = 0; r < 8; ++r) {
      int m = m0 + rbase + r;
      if (m >= M) continue;
      float v = acc[i][r] + bv;
      if (relu) v = fmaxf(v, 0.f);
      size_t o = (size_t)m * N + n;
      if (outF) outF[o] = v;
      if (outH) outH[o] = (_Float16)v;
    }
  }
}

// ---------------------------------------------------------------------------
// scores[b][t] = feats[b][t][:] . attn_v   (padded stride 512)
// ---------------------------------------------------------------------------
__global__ void scores_kernel(const _Float16* __restrict__ feats,
                              const float* __restrict__ av, float* __restrict__ scores) {
  int i = blockIdx.x * blockDim.x + threadIdx.x;
  if (i >= 4 * 499) return;
  int b = i / 499, t = i % 499;
  const _Float16* f = feats + ((size_t)b * 499 + t) * 768;
  float acc = 0.f;
  for (int c = 0; c < 768; ++c) acc = fmaf((float)f[c], av[c], acc);
  scores[b * 512 + t] = acc;
}

// ---------------------------------------------------------------------------
// Per-segment masked-softmax attention pooling + pos_emb. One block per (b,s).
// ---------------------------------------------------------------------------
__global__ void pool_kernel(const _Float16* __restrict__ feats, const float* __restrict__ scores,
                            const int* __restrict__ seg_bounds, const int* __restrict__ n_segs,
                            const float* __restrict__ pos_emb,
                            float* __restrict__ tok, _Float16* __restrict__ tok_h) {
  const int b = blockIdx.x >> 6;
  const int s = blockIdx.x & 63;
  const int tid = threadIdx.x;
  __shared__ float red[256];
  __shared__ float wts[512];
  const int ns = n_segs[b];
  size_t obase = ((size_t)b * 64 + s) * 768;
  if (s >= ns) {  // padded segment -> zeros + pos_emb
    for (int c = tid; c < 768; c += 256) {
      float v = pos_emb[s * 768 + c];
      tok[obase + c] = v;
      tok_h[obase + c] = (_Float16)v;
    }
    return;
  }
  int st = seg_bounds[b * 65 + s];
  int en = seg_bounds[b * 65 + s + 1];
  st = min(max(st, 0), 498);
  en = min(max(en, 0), 499);
  en = max(st + 1, en);
  float lm = -1e30f;
  for (int t = st + tid; t < en; t += 256) lm = fmaxf(lm, scores[b * 512 + t]);
  red[tid] = lm; __syncthreads();
  for (int off = 128; off > 0; off >>= 1) {
    if (tid < off) red[tid] = fmaxf(red[tid], red[tid + off]);
    __syncthreads();
  }
  float m = red[0]; __syncthreads();
  float ls = 0.f;
  for (int t = st + tid; t < en; t += 256) {
    float e = __expf(scores[b * 512 + t] - m);
    wts[t - st] = e;
    ls += e;
  }
  red[tid] = ls; __syncthreads();
  for (int off = 128; off > 0; off >>= 1) {
    if (tid < off) red[tid] += red[tid + off];
    __syncthreads();
  }
  float den = red[0];
  float inv = (den > 0.f) ? 1.f / den : 1.f;
  for (int c = tid; c < 768; c += 256) {
    float acc = 0.f;
    for (int t = st; t < en; ++t)
      acc = fmaf(wts[t - st], (float)feats[((size_t)b * 499 + t) * 768 + c], acc);
    float v = acc * inv + pos_emb[s * 768 + c];
    tok[obase + c] = v;
    tok_h[obase + c] = (_Float16)v;
  }
}

// ---------------------------------------------------------------------------
// Attention: one block per (b,h), 64 threads (one per query). K/V/P tiles in LDS.
// ---------------------------------------------------------------------------
__global__ void attn_kernel(const float* __restrict__ qkv, const int* __restrict__ n_segs,
                            _Float16* __restrict__ o_h) {
  const int b = blockIdx.x / 12;
  const int h = blockIdx.x % 12;
  const int sq = threadIdx.x;
  __shared__ float Ks[64][64];
  __shared__ float Vs[64][64];
  __shared__ float Ps[64][64];
  for (int idx = threadIdx.x; idx < 4096; idx += 64) {
    int sk = idx >> 6, d = idx & 63;
    size_t base = ((size_t)b * 64 + sk) * 2304 + h * 64;
    Ks[sk][d] = qkv[base + 768 + d];
    Vs[sk][d] = qkv[base + 1536 + d];
  }
  __syncthreads();
  float q[64];
  size_t qb = ((size_t)b * 64 + sq) * 2304 + h * 64;
#pragma unroll
  for (int d = 0; d < 64; ++d) q[d] = qkv[qb + d];
  const int ns = n_segs[b];
  float m = -1e30f;
  for (int j = 0; j < 64; ++j) {
    float acc = 0.f;
#pragma unroll
    for (int d = 0; d < 64; ++d) acc = fmaf(q[d], Ks[j][d], acc);
    acc = (j < ns) ? acc * 0.125f : -1e9f;   // key padding mask
    Ps[sq][j] = acc;
    m = fmaxf(m, acc);
  }
  float den = 0.f;
  for (int j = 0; j < 64; ++j) {
    float e = __expf(Ps[sq][j] - m);
    Ps[sq][j] = e;
    den += e;
  }
  float inv = 1.f / den;
  for (int d = 0; d < 64; ++d) {
    float acc = 0.f;
#pragma unroll
    for (int j = 0; j < 64; ++j) acc = fmaf(Ps[sq][j], Vs[j][d], acc);
    o_h[((size_t)b * 64 + sq) * 768 + h * 64 + d] = (_Float16)(acc * inv);
  }
}

// ---------------------------------------------------------------------------
// Residual + LayerNorm (delta==null -> plain LN). One block per row, D=768.
// ---------------------------------------------------------------------------
__global__ void addln_kernel(const float* __restrict__ base, const float* __restrict__ delta,
                             const float* __restrict__ gamma, const float* __restrict__ beta,
                             float* __restrict__ outF, _Float16* __restrict__ outH) {
  const int row = blockIdx.x;
  const int tid = threadIdx.x;
  __shared__ float red[256];
  const float* bp = base + (size_t)row * 768;
  const float* dp = delta ? delta + (size_t)row * 768 : nullptr;
  float hv[3];
  float s = 0.f;
#pragma unroll
  for (int i = 0; i < 3; ++i) {
    int c = tid + i * 256;
    float v = bp[c] + (dp ? dp[c] : 0.f);
    hv[i] = v;
    s += v;
  }
  red[tid] = s; __syncthreads();
  for (int off = 128; off > 0; off >>= 1) {
    if (tid < off) red[tid] += red[tid + off];
    __syncthreads();
  }
  float mean = red[0] * (1.f / 768.f); __syncthreads();
  float vs = 0.f;
#pragma unroll
  for (int i = 0; i < 3; ++i) { float d0 = hv[i] - mean; vs += d0 * d0; }
  red[tid] = vs; __syncthreads();
  for (int off = 128; off > 0; off >>= 1) {
    if (tid < off) red[tid] += red[tid + off];
    __syncthreads();
  }
  float rstd = rsqrtf(red[0] * (1.f / 768.f) + 1e-5f);
#pragma unroll
  for (int i = 0; i < 3; ++i) {
    int c = tid + i * 256;
    float v = (hv[i] - mean) * rstd * gamma[c] + beta[c];
    size_t o = (size_t)row * 768 + c;
    if (outF) outF[o] = v;
    if (outH) outH[o] = (_Float16)v;
  }
}

__global__ void mask_kernel(const int* __restrict__ n_segs, float* __restrict__ out) {
  int i = threadIdx.x;  // 256 = B*SMAX
  int b = i >> 6, s = i & 63;
  out[i] = (s >= n_segs[b]) ? 1.0f : 0.0f;
}

// ---------------------------------------------------------------------------
extern "C" void kernel_launch(void* const* d_in, const int* in_sizes, int n_in,
                              void* d_out, int out_size, void* d_ws, size_t ws_size,
                              hipStream_t stream) {
  (void)in_sizes; (void)n_in; (void)ws_size;
  const float* x          = (const float*)d_in[0];
  const int*   seg_bounds = (const int*)d_in[1];
  const int*   n_segs     = (const int*)d_in[2];
  const float* conv_w[7]; const float* conv_b[7];
  for (int i = 0; i < 7; ++i) {
    conv_w[i] = (const float*)d_in[3 + 2 * i];
    conv_b[i] = (const float*)d_in[4 + 2 * i];
  }
  const float* attn_v  = (const float*)d_in[17];
  const float* pos_emb = (const float*)d_in[18];
  const float* in_w  = (const float*)d_in[19];
  const float* in_b  = (const float*)d_in[20];
  const float* out_w = (const float*)d_in[21];
  const float* out_b = (const float*)d_in[22];
  const float* ff1_w = (const float*)d_in[23];
  const float* ff1_b = (const float*)d_in[24];
  const float* ff2_w = (const float*)d_in[25];
  const float* ff2_b = (const float*)d_in[26];
  const float* ln1_w = (const float*)d_in[27];
  const float* ln1_b = (const float*)d_in[28];
  const float* ln2_w = (const float*)d_in[29];
  const float* ln2_b = (const float*)d_in[30];
  const float* fln_w = (const float*)d_in[31];
  const float* fln_b = (const float*)d_in[32];
  const float* proj_w = (const float*)d_in[33];
  const float* proj_b = (const float*)d_in[34];
  float* outp = (float*)d_out;

  // ---- workspace carve (256B aligned) ----
  char* ws = (char*)d_ws;
  size_t off = 0;
  auto carve = [&](size_t bytes) -> void* {
    void* p = ws + off;
    off = (off + bytes + 255) & ~(size_t)255;
    return p;
  };
  _Float16* actA  = (_Float16*)carve((size_t)4 * 31999 * 512 * 2);  // 131 MB
  _Float16* actB  = (_Float16*)carve((size_t)4 * 15999 * 512 * 2);  //  66 MB
  _Float16* wbuf  = (_Float16*)carve((size_t)4 << 20);              // weight f16 scratch
  float*    scr   = (float*)carve((size_t)4 * 512 * 4);
  float*    tok   = (float*)carve((size_t)256 * 768 * 4);
  _Float16* tok_h = (_Float16*)carve((size_t)256 * 768 * 2);
  float*    qkv   = (float*)carve((size_t)256 * 2304 * 4);
  _Float16* o_h   = (_Float16*)carve((size_t)256 * 768 * 2);
  float*    tmp   = (float*)carve((size_t)256 * 768 * 4);
  _Float16* ff_h  = (_Float16*)carve((size_t)256 * 2048 * 2);

  auto cast = [&](const float* src, _Float16* dst, int n) {
    cast_f32_f16_kernel<<<(n + 255) / 256, 256, 0, stream>>>(src, dst, n);
  };

  // ---- conv feature extractor ----
  conv0_kernel<<<dim3(31999, 4), 256, 0, stream>>>(x, conv_w[0], conv_b[0], actA);

  struct CS { int oc, ic, k, s, Lin, Lout; };
  const CS cs[7] = {{512, 1, 10, 5, 160000, 31999}, {512, 512, 3, 2, 31999, 15999},
                    {512, 512, 3, 2, 15999, 7999},  {512, 512, 3, 2, 7999, 3999},
                    {512, 512, 2, 2, 3999, 1999},   {512, 512, 2, 2, 1999, 999},
                    {768, 512, 2, 2, 999, 499}};
  _Float16* cur = actA;
  _Float16* nxt = actB;
  for (int l = 1; l < 7; ++l) {
    int n = cs[l].oc * cs[l].ic * cs[l].k;
    convw_cast_kernel<<<(n + 255) / 256, 256, 0, stream>>>(conv_w[l], wbuf, cs[l].oc,
                                                           cs[l].ic, cs[l].k);
    dim3 grid((cs[l].Lout + 511) / 512, cs[l].oc / 16, 4);   // 64 t per wave, 8 waves/block
    conv_wmma_kernel<<<grid, 256, 0, stream>>>(cur, wbuf, conv_b[l], nxt, cs[l].Lin,
                                               cs[l].Lout, cs[l].ic, cs[l].oc, cs[l].k,
                                               cs[l].s);
    _Float16* t2 = cur; cur = nxt; nxt = t2;
  }
  const _Float16* feats = cur;  // [4,499,768] f16, ends in actA

  // ---- segment attention pooling ----
  scores_kernel<<<(4 * 499 + 255) / 256, 256, 0, stream>>>(feats, attn_v, scr);
  pool_kernel<<<256, 256, 0, stream>>>(feats, scr, seg_bounds, n_segs, pos_emb, tok, tok_h);

  // ---- transformer encoder, 12 layers ----
  for (int l = 0; l < 12; ++l) {
    cast(in_w + (size_t)l * 2304 * 768, wbuf, 2304 * 768);
    linear_wmma_kernel<<<dim3(5, 16), 256, 0, stream>>>(tok_h, wbuf, in_b + l * 2304,
                                                        qkv, nullptr, 256, 2304, 768, 0);
    attn_kernel<<<48, 64, 0, stream>>>(qkv, n_segs, o_h);
    cast(out_w + (size_t)l * 768 * 768, wbuf, 768 * 768);
    linear_wmma_kernel<<<dim3(2, 16), 256, 0, stream>>>(o_h, wbuf, out_b + l * 768,
                                                        tmp, nullptr, 256, 768, 768, 0);
    addln_kernel<<<256, 256, 0, stream>>>(tok, tmp, ln1_w + l * 768, ln1_b + l * 768,
                                          tok, tok_h);
    cast(ff1_w + (size_t)l * 2048 * 768, wbuf, 2048 * 768);
    linear_wmma_kernel<<<dim3(4, 16), 256, 0, stream>>>(tok_h, wbuf, ff1_b + l * 2048,
                                                        nullptr, ff_h, 256, 2048, 768, 1);
    cast(ff2_w + (size_t)l * 768 * 2048, wbuf, 768 * 2048);
    linear_wmma_kernel<<<dim3(2, 16), 256, 0, stream>>>(ff_h, wbuf, ff2_b + l * 768,
                                                        tmp, nullptr, 256, 768, 2048, 0);
    addln_kernel<<<256, 256, 0, stream>>>(tok, tmp, ln2_w + l * 768, ln2_b + l * 768,
                                          tok, tok_h);
  }

  // ---- final LN + projection ----
  addln_kernel<<<256, 256, 0, stream>>>(tok, nullptr, fln_w, fln_b, nullptr, tok_h);
  cast(proj_w, wbuf, 100 * 768);
  linear_wmma_kernel<<<dim3(1, 16), 256, 0, stream>>>(tok_h, wbuf, proj_b, outp, nullptr,
                                                      256, 100, 768, 0);
  if (out_size >= 25600 + 256)
    mask_kernel<<<1, 256, 0, stream>>>(n_segs, outp + 25600);
}
